// PullbackOneForm_52123723104866
// MI455X (gfx1250) — matrix operand order repaired and checked by reading, not compiled, not run on hardware
//
#include <hip/hip_runtime.h>

typedef __attribute__((ext_vector_type(16))) __bf16 v16bf;
typedef __attribute__((ext_vector_type(8)))  float  v8f;
typedef __attribute__((ext_vector_type(4)))  float  v4f;

constexpr int H   = 512;   // hidden width
constexpr int MT  = 32;    // rows per block
constexpr int NT  = 256;   // threads per block (8 waves of 32)

union Frag {
    v4f   q[2];
    v16bf v;
};

// Branch-free tanh: 1 - 2/(exp(2x)+1), exp via v_exp_f32, div via v_rcp_f32.
// Saturates exactly to +/-1 for large |x| (inf -> rcp 0, 0 -> rcp 1).
__device__ inline float fast_tanh(float x) {
    const float e = __builtin_amdgcn_exp2f(x * 2.8853900817779268f);  // exp(2x)
    const float r = __builtin_amdgcn_rcpf(e + 1.0f);
    return fmaf(-2.0f, r, 1.0f);
}

// ---- WMMA fragment loaders -------------------------------------------------
// A (16x32 bf16, MxK), act stored row-major in LDS with row stride H.
// lane<16 : M=lane,    K = {k0..k0+7, k0+16..k0+23}
// lane>=16: M=lane-16, K = {k0+8..k0+15, k0+24..k0+31}
__device__ inline v16bf load_a(const __bf16* base, int lane, int k0) {
    const int r  = lane & 15;
    const int kk = k0 + ((lane >> 4) << 3);
    const __bf16* p = base + r * H + kk;
    Frag f;
    f.q[0] = *(const v4f*)(p);        // K chunk 0..7 (or 8..15)
    f.q[1] = *(const v4f*)(p + 16);   // K chunk 16..23 (or 24..31)
    return f.v;
}

// B (32x16 bf16, KxN), weights pre-transposed: Bt[n][k] = W[k][n], row stride H.
// lane<16 : N=lane,    K = {k0..k0+7, k0+16..k0+23}
// lane>=16: N=lane-16, K = {k0+8..k0+15, k0+24..k0+31}
__device__ inline v16bf load_b(const __bf16* __restrict__ Bt, int lane, int col0, int k0) {
    const int n  = col0 + (lane & 15);
    const int kk = k0 + ((lane >> 4) << 3);
    const __bf16* p = Bt + n * H + kk;
    Frag f;
    f.q[0] = *(const v4f*)(p);        // K chunk 0..7 (or 8..15)
    f.q[1] = *(const v4f*)(p + 16);   // K chunk 16..23 (or 24..31)
    return f.v;
}

__device__ inline v8f wmma_bf16(v16bf a, v16bf b, v8f c) {
    return __builtin_amdgcn_wmma_f32_16x16x32_bf16(
        /*neg_a=*/false, a, /*neg_b=*/false, b,
        /*c_mod=*/(short)0, c, /*reuse_a=*/false, /*reuse_b=*/false);
}

// ---- weight prepass: fp32 row-major (K x N) -> bf16 transposed (N x K) -----
__global__ void convert_weights(const float* __restrict__ dW2,
                                const float* __restrict__ oW2,
                                __bf16* __restrict__ dW2t,
                                __bf16* __restrict__ oW2t) {
    int idx = blockIdx.x * blockDim.x + threadIdx.x;   // 0 .. H*H-1
    if (idx >= H * H) return;
    int n = idx >> 9;          // output row (= weight col)
    int k = idx & (H - 1);     // output col (= weight row)
    dW2t[idx] = (__bf16)dW2[k * H + n];
    oW2t[idx] = (__bf16)oW2[k * H + n];
}

// ---- fused kernel ----------------------------------------------------------
struct alignas(16) SMem {
    __bf16 b[6][MT][H];        // activation ping-pong buffers (192 KB)
    float  part[MT][8][9];     // partial reductions for 512->3 heads
    float  xyzJ[MT][9];        // per-row: xyz[3], J0[3], J1[3]
    float  uvs[MT][2];         // staged uv for this tile
};

__global__ __launch_bounds__(NT)
void PullbackOneForm_kernel(const float* __restrict__ uv,
                            const float* __restrict__ dW1, const float* __restrict__ db1,
                            const __bf16* __restrict__ dW2t, const float* __restrict__ db2,
                            const float* __restrict__ dW3, const float* __restrict__ db3,
                            const float* __restrict__ oW1, const float* __restrict__ ob1,
                            const __bf16* __restrict__ oW2t, const float* __restrict__ ob2,
                            const float* __restrict__ oW3, const float* __restrict__ ob3,
                            float* __restrict__ out, int N) {
    __shared__ SMem sm;

    const int tid   = threadIdx.x;
    const int wave  = tid >> 5;
    const int lane  = tid & 31;
    const int gRow0 = blockIdx.x * MT;

    // ---------------- stage uv ---------------------------------------------
    if (tid < MT * 2) {
        const int row = tid >> 1, c = tid & 1;
        int gr = gRow0 + row; if (gr >= N) gr = N - 1;
        sm.uvs[row][c] = uv[gr * 2 + c];
    }
    __syncthreads();

    // ---------------- layer d1: 2 -> H, tanh, plus tangents -----------------
    for (int idx = tid; idx < MT * H; idx += NT) {
        const int row = idx >> 9;
        const int col = idx & (H - 1);
        const float u  = sm.uvs[row][0];
        const float v  = sm.uvs[row][1];
        const float w0 = dW1[col];
        const float w1 = dW1[H + col];
        const float z  = fmaf(u, w0, fmaf(v, w1, db1[col]));
        const float h  = fast_tanh(z);
        const float s  = 1.0f - h * h;
        sm.b[0][row][col] = (__bf16)h;          // h1
        sm.b[1][row][col] = (__bf16)(s * w0);   // dh1 / du
        sm.b[2][row][col] = (__bf16)(s * w1);   // dh1 / dv
    }
    __syncthreads();

    // ------- layer d2: H -> H GEMM, primal + 2 tangents, both row halves ----
    // Each wave owns 4 column tiles and computes all 32 rows against a single
    // B fragment per k-step: 6 WMMAs per B load (halves L2 weight traffic).
    {
        const int jbase = wave * 4;
        for (int jt = 0; jt < 4; ++jt) {
            const int col0 = (jbase + jt) * 16;
            v8f accP[2] = {{}, {}}, accT0[2] = {{}, {}}, accT1[2] = {{}, {}};
            #pragma unroll 4
            for (int k0 = 0; k0 < H; k0 += 32) {
                const v16bf b = load_b(dW2t, lane, col0, k0);
                accP[0]  = wmma_bf16(load_a(&sm.b[0][0][0],  lane, k0), b, accP[0]);
                accP[1]  = wmma_bf16(load_a(&sm.b[0][16][0], lane, k0), b, accP[1]);
                accT0[0] = wmma_bf16(load_a(&sm.b[1][0][0],  lane, k0), b, accT0[0]);
                accT0[1] = wmma_bf16(load_a(&sm.b[1][16][0], lane, k0), b, accT0[1]);
                accT1[0] = wmma_bf16(load_a(&sm.b[2][0][0],  lane, k0), b, accT1[0]);
                accT1[1] = wmma_bf16(load_a(&sm.b[2][16][0], lane, k0), b, accT1[1]);
            }
            const int col  = col0 + (lane & 15);
            const float bs = db2[col];
            #pragma unroll
            for (int rh = 0; rh < 2; ++rh) {
                #pragma unroll
                for (int r = 0; r < 8; ++r) {
                    const int row = rh * 16 + r + ((lane >> 4) << 3);
                    const float h = fast_tanh(accP[rh][r] + bs);
                    const float s = 1.0f - h * h;
                    sm.b[3][row][col] = (__bf16)h;                  // h2
                    sm.b[4][row][col] = (__bf16)(s * accT0[rh][r]); // dh2/du
                    sm.b[5][row][col] = (__bf16)(s * accT1[rh][r]); // dh2/dv
                }
            }
        }
    }
    __syncthreads();

    // ---------------- layer d3: H -> 3 (xyz, J0, J1) ------------------------
    {
        const int row = tid >> 3;
        const int sl  = tid & 7;
        float acc[9] = {};
        for (int k = sl * 64; k < sl * 64 + 64; ++k) {
            const float w0 = dW3[k * 3 + 0];
            const float w1 = dW3[k * 3 + 1];
            const float w2 = dW3[k * 3 + 2];
            const float hp = (float)sm.b[3][row][k];
            const float t0 = (float)sm.b[4][row][k];
            const float t1 = (float)sm.b[5][row][k];
            acc[0] = fmaf(hp, w0, acc[0]); acc[1] = fmaf(hp, w1, acc[1]); acc[2] = fmaf(hp, w2, acc[2]);
            acc[3] = fmaf(t0, w0, acc[3]); acc[4] = fmaf(t0, w1, acc[4]); acc[5] = fmaf(t0, w2, acc[5]);
            acc[6] = fmaf(t1, w0, acc[6]); acc[7] = fmaf(t1, w1, acc[7]); acc[8] = fmaf(t1, w2, acc[8]);
        }
        #pragma unroll
        for (int i = 0; i < 9; ++i) sm.part[row][sl][i] = acc[i];
    }
    __syncthreads();
    for (int o = tid; o < MT * 9; o += NT) {
        const int row = o / 9, c = o % 9;
        float s = 0.0f;
        #pragma unroll
        for (int sl = 0; sl < 8; ++sl) s += sm.part[row][sl][c];
        if (c < 3) s += db3[c];      // bias only on xyz, not Jacobian
        sm.xyzJ[row][c] = s;
    }
    __syncthreads();

    // ---------------- layer o1: 3 -> H, tanh --------------------------------
    for (int idx = tid; idx < MT * H; idx += NT) {
        const int row = idx >> 9;
        const int col = idx & (H - 1);
        const float x = sm.xyzJ[row][0], y = sm.xyzJ[row][1], zc = sm.xyzJ[row][2];
        float z = ob1[col];
        z = fmaf(x,  oW1[col],         z);
        z = fmaf(y,  oW1[H + col],     z);
        z = fmaf(zc, oW1[2 * H + col], z);
        sm.b[0][row][col] = (__bf16)fast_tanh(z);   // g1
    }
    __syncthreads();

    // ---------------- layer o2: H -> H GEMM (both row halves per wave) ------
    {
        const int jbase = wave * 4;
        for (int jt = 0; jt < 4; ++jt) {
            const int col0 = (jbase + jt) * 16;
            v8f acc[2] = {{}, {}};
            #pragma unroll 4
            for (int k0 = 0; k0 < H; k0 += 32) {
                const v16bf b = load_b(oW2t, lane, col0, k0);
                acc[0] = wmma_bf16(load_a(&sm.b[0][0][0],  lane, k0), b, acc[0]);
                acc[1] = wmma_bf16(load_a(&sm.b[0][16][0], lane, k0), b, acc[1]);
            }
            const int col  = col0 + (lane & 15);
            const float bs = ob2[col];
            #pragma unroll
            for (int rh = 0; rh < 2; ++rh) {
                #pragma unroll
                for (int r = 0; r < 8; ++r) {
                    const int row = rh * 16 + r + ((lane >> 4) << 3);
                    sm.b[1][row][col] = (__bf16)fast_tanh(acc[rh][r] + bs);   // g2
                }
            }
        }
    }
    __syncthreads();

    // ---------------- layer o3: H -> 3, then 1-form contraction -------------
    {
        const int row = tid >> 3;
        const int sl  = tid & 7;
        float a0 = 0.0f, a1 = 0.0f, a2 = 0.0f;
        for (int k = sl * 64; k < sl * 64 + 64; ++k) {
            const float g = (float)sm.b[1][row][k];
            a0 = fmaf(g, oW3[k * 3 + 0], a0);
            a1 = fmaf(g, oW3[k * 3 + 1], a1);
            a2 = fmaf(g, oW3[k * 3 + 2], a2);
        }
        sm.part[row][sl][0] = a0; sm.part[row][sl][1] = a1; sm.part[row][sl][2] = a2;
    }
    __syncthreads();
    if (tid < MT) {
        const int row = tid;
        float f0 = ob3[0], f1 = ob3[1], f2 = ob3[2];
        #pragma unroll
        for (int sl = 0; sl < 8; ++sl) {
            f0 += sm.part[row][sl][0];
            f1 += sm.part[row][sl][1];
            f2 += sm.part[row][sl][2];
        }
        const float p = f0 * sm.xyzJ[row][3] + f1 * sm.xyzJ[row][4] + f2 * sm.xyzJ[row][5];
        const float q = f0 * sm.xyzJ[row][6] + f1 * sm.xyzJ[row][7] + f2 * sm.xyzJ[row][8];
        const int gr = gRow0 + row;
        if (gr < N) {
            out[gr * 2 + 0] = p;
            out[gr * 2 + 1] = q;
        }
    }
}

extern "C" void kernel_launch(void* const* d_in, const int* in_sizes, int n_in,
                              void* d_out, int out_size, void* d_ws, size_t ws_size,
                              hipStream_t stream) {
    const float* uv  = (const float*)d_in[0];
    const float* dW1 = (const float*)d_in[1];
    const float* db1 = (const float*)d_in[2];
    const float* dW2 = (const float*)d_in[3];
    const float* db2 = (const float*)d_in[4];
    const float* dW3 = (const float*)d_in[5];
    const float* db3 = (const float*)d_in[6];
    const float* oW1 = (const float*)d_in[7];
    const float* ob1 = (const float*)d_in[8];
    const float* oW2 = (const float*)d_in[9];
    const float* ob2 = (const float*)d_in[10];
    const float* oW3 = (const float*)d_in[11];
    const float* ob3 = (const float*)d_in[12];

    const int N = in_sizes[0] / 2;

    __bf16* dW2t = (__bf16*)d_ws;          // H*H bf16 = 512 KB
    __bf16* oW2t = dW2t + H * H;           // + 512 KB (ws usage: 1 MB)

    convert_weights<<<dim3((H * H + NT - 1) / NT), dim3(NT), 0, stream>>>(dW2, oW2, dW2t, oW2t);

    const int nBlocks = (N + MT - 1) / MT;
    PullbackOneForm_kernel<<<dim3(nBlocks), dim3(NT), 0, stream>>>(
        uv, dW1, db1, dW2t, db2, dW3, db3, oW1, ob1, oW2t, ob2, oW3, ob3,
        (float*)d_out, N);
}
